// RAFTDVC_81982335746194
// MI455X (gfx1250) — matrix operand order repaired and checked by reading, not compiled, not run on hardware
//
#include <hip/hip_runtime.h>
#include <math.h>

#define ACT_NONE 0
#define ACT_RELU 1
#define ACT_SIGMOID 2
#define ACT_TANH 3

typedef __attribute__((ext_vector_type(16))) __bf16 v16bf;
typedef __attribute__((ext_vector_type(8)))  float  v8f;

namespace raftdvc {

constexpr int HFULL = 96;
constexpr int NFULL = HFULL*HFULL*HFULL;   // 884736
constexpr int HT    = 12;
constexpr int NV    = HT*HT*HT;            // 1728
constexpr int FDIM  = 128;
constexpr int ITERS = 12;
constexpr int KOFF  = 729;                 // (2*4+1)^3
constexpr int CORR_PLANES = 4*KOFF;        // 2916
constexpr int TABMAX = 6912;               // max Ktot with kvol>1 (fh2: 256*27)

__device__ __forceinline__ unsigned short f2bf(float f){
  union { float f; unsigned u; } c; c.f = f;
  return (unsigned short)((c.u + 0x7fffu + ((c.u >> 16) & 1u)) >> 16);
}

// ===================== WMMA implicit-GEMM 3D conv / GEMM =====================
// C[m][co] = act( scale * sum_k A(m,k)*B(k,co) + bias[co] )
// A: bf16 activations [Cin][Di][Hi][Wi]; k = (ci,kz,ky,kx); zero outside volume.
// B: bf16 repacked weights [Kpad][Npad] (zero padded).
// Block = 4 waves (128 thr). Wave w owns M-tile (bx*4+w), all waves share the
// 32-wide N-panel blockIdx.y: B staged in LDS (col-major, double buffered),
// k->(ci,kz,ky,kx) decomposition precomputed once per block into LDS.
// Each wave runs TWO 16x16x32 WMMAs per K-step (2 N-subtiles share one A frag).
__global__ __launch_bounds__(128)
void k_conv3d_wmma(const unsigned short* __restrict__ A,
                   const unsigned short* __restrict__ Bw,
                   const float* __restrict__ bias,
                   float* __restrict__ outF,
                   unsigned short* __restrict__ outB,
                   int M, int Ho, int Wo,
                   int Di, int Hi, int Wi,
                   int kvol, int khw, int kw, int stride,
                   int pd, int ph, int pw,
                   int Ktot, int Kpad, int Cout, int Npad,
                   float scale, int act)
{
  __shared__ int tab[TABMAX];
  __shared__ unsigned short bsh[2][32*32];   // [buf][colLocal*32 + row]

  const int tid  = threadIdx.x;
  const int wave = tid >> 5;
  const int lane = tid & 31;
  const int lh   = lane >> 4;
  const int l15  = lane & 15;

  // ---- block-wide: decomposition table (divisions once per block) ----
  if (kvol > 1) {
    for (int k = tid; k < Ktot && k < TABMAX; k += 128) {
      int ci = k / kvol; int r = k - ci*kvol;
      int kz = r / khw;  int r2 = r - kz*khw;
      int ky = r2 / kw;  int kx = r2 - ky*kw;
      tab[k] = (ci << 12) | (kz << 8) | (ky << 4) | kx;
    }
  }

  const int Mtiles = M >> 4;
  int tileM = blockIdx.x*4 + wave;
  const bool activeM = (tileM < Mtiles);
  if (!activeM) tileM = Mtiles - 1;          // compute redundantly, skip store
  const int nbase = blockIdx.y * 32;

  // A row (M) handled by this lane
  const int m  = tileM*16 + l15;
  const int ox = m % Wo; int t = m / Wo;
  const int oy = t % Ho; const int oz = t / Ho;
  const int bz = oz*stride - pd, by = oy*stride - ph, bx = ox*stride - pw;

  // K offsets per documented 16-bit A layout (16x32, wave32)
  int klA[16];
#pragma unroll
  for (int v = 0; v < 8; ++v)
#pragma unroll
    for (int e = 0; e < 2; ++e)
      klA[v*2+e] = ((v < 4) ? (v*2+e) : (16 + (v-4)*2 + e)) + 8*lh;

  // ---- B panel staging: [32 rows K][32 cols N] -> LDS col-major ----
  const int frow = tid >> 2;            // 0..31
  const int fcol = (tid & 3) << 3;      // 0,8,16,24
  auto fillB = [&](int buf, int kb) {
    unsigned short tmp[8];
    if (nbase + fcol + 8 <= Npad) {
      const unsigned short* src = Bw + (size_t)(kb + frow)*Npad + nbase + fcol;
      *(uint4*)tmp = *(const uint4*)src;                 // 16B vector load
      if (kb + 64 < Kpad)                                // -> global_prefetch_b8
        __builtin_prefetch(src + (size_t)32*Npad, 0, 1);
    } else {
#pragma unroll
      for (int j = 0; j < 8; ++j) tmp[j] = 0;
    }
#pragma unroll
    for (int j = 0; j < 8; ++j) bsh[buf][(fcol + j)*32 + frow] = tmp[j];
  };

  v8f acc0 = {0.f,0.f,0.f,0.f,0.f,0.f,0.f,0.f};
  v8f acc1 = {0.f,0.f,0.f,0.f,0.f,0.f,0.f,0.f};

  fillB(0, 0);

  for (int kb = 0; kb < Kpad; kb += 32) {
    __syncthreads();                       // fill(buf) + tab visible
    const int buf = (kb >> 5) & 1;
    if (kb + 32 < Kpad) fillB(buf ^ 1, kb + 32);

    // ---- A fragment: 16 gathers, table-driven decomposition ----
    union { v16bf v; unsigned short u[16]; } a;
#pragma unroll
    for (int i = 0; i < 16; ++i) {
      int k = kb + klA[i];
      unsigned short av = 0;
      if (k < Ktot) {
        int ci, iz, iy, ix;
        if (kvol == 1) { ci = k; iz = bz; iy = by; ix = bx; }
        else {
          int e = tab[k];
          ci = e >> 12;
          iz = bz + ((e >> 8) & 15);
          iy = by + ((e >> 4) & 15);
          ix = bx + (e & 15);
        }
        if ((unsigned)iz < (unsigned)Di && (unsigned)iy < (unsigned)Hi &&
            (unsigned)ix < (unsigned)Wi)
          av = A[((size_t)ci*Di + iz)*Hi*Wi + (size_t)iy*Wi + ix];
      }
      a.u[i] = av;
    }

    // ---- B fragments from LDS: two 16B chunks per lane (k=8h.. and 16+8h..) -> 2x ds_load_b128 ----
    union { v16bf v; unsigned short u[16]; } b0, b1;
    {
      const unsigned short* p0 = &bsh[buf][(l15      )*32 + lh*8];
      const unsigned short* p1 = &bsh[buf][(16 + l15 )*32 + lh*8];
      *(uint4*)&b0.u[0] = *(const uint4*)p0;
      *(uint4*)&b0.u[8] = *(const uint4*)(p0 + 16);
      *(uint4*)&b1.u[0] = *(const uint4*)p1;
      *(uint4*)&b1.u[8] = *(const uint4*)(p1 + 16);
    }

    acc0 = __builtin_amdgcn_wmma_f32_16x16x32_bf16(false, a.v, false, b0.v,
                                                   (short)0, acc0, false, false);
    acc1 = __builtin_amdgcn_wmma_f32_16x16x32_bf16(false, a.v, false, b1.v,
                                                   (short)0, acc1, false, false);
  }

  if (!activeM) return;
#pragma unroll
  for (int sub = 0; sub < 2; ++sub) {
    int co = nbase + sub*16 + l15;
    if (co >= Cout) continue;
    float bv = bias ? bias[co] : 0.f;
#pragma unroll
    for (int v = 0; v < 8; ++v) {
      int mr = tileM*16 + v + 8*lh;       // C layout: VGPR v -> M=v (+8 upper half)
      float val = (sub ? acc1[v] : acc0[v])*scale + bv;
      if (act == ACT_RELU)         val = fmaxf(val, 0.f);
      else if (act == ACT_SIGMOID) val = 1.f/(1.f+__expf(-val));
      else if (act == ACT_TANH)    val = tanhf(val);
      size_t o = (size_t)co*M + mr;
      if (outF) outF[o] = val;
      if (outB) outB[o] = f2bf(val);
    }
  }
}

// ===================== weight repack OIDHW f32 -> [Kpad][Npad] bf16 ==========
__global__ void k_repack(const float* __restrict__ w, unsigned short* __restrict__ o,
                         int Cout, int Cin, int kd, int kh, int kw, int Kpad, int Npad)
{
  size_t idx = (size_t)blockIdx.x*blockDim.x + threadIdx.x;
  size_t tot = (size_t)Kpad*Npad;
  if (idx >= tot) return;
  int k  = (int)(idx / Npad);
  int co = (int)(idx - (size_t)k*Npad);
  int kvol = kd*kh*kw, khw = kh*kw, Ktot = Cin*kvol;
  float v = 0.f;
  if (k < Ktot && co < Cout) {
    int ci = k / kvol; int r = k - ci*kvol;
    int kz = r / khw;  int r2 = r - kz*khw;
    int ky = r2 / kw;  int kx = r2 - ky*kw;
    v = w[((((size_t)co*Cin + ci)*kd + kz)*kh + ky)*kw + kx];
  }
  o[idx] = f2bf(v);
}

// ===================== min/max + normalize ===================================
__global__ void k_minmax_part(const float* __restrict__ a, const float* __restrict__ b,
                              int n, float* __restrict__ pmin, float* __restrict__ pmax)
{
  __shared__ float smn[256], smx[256];
  float mn = 3.4e38f, mx = -3.4e38f;
  for (long long i = (long long)blockIdx.x*blockDim.x + threadIdx.x;
       i < 2LL*n; i += (long long)gridDim.x*blockDim.x) {
    float v = (i < n) ? a[i] : b[i - n];
    mn = fminf(mn, v); mx = fmaxf(mx, v);
  }
  smn[threadIdx.x] = mn; smx[threadIdx.x] = mx; __syncthreads();
  for (int s = 128; s > 0; s >>= 1) {
    if ((int)threadIdx.x < s) {
      smn[threadIdx.x] = fminf(smn[threadIdx.x], smn[threadIdx.x+s]);
      smx[threadIdx.x] = fmaxf(smx[threadIdx.x], smx[threadIdx.x+s]);
    }
    __syncthreads();
  }
  if (threadIdx.x == 0) { pmin[blockIdx.x] = smn[0]; pmax[blockIdx.x] = smx[0]; }
}

__global__ void k_minmax_final(const float* __restrict__ pmin, const float* __restrict__ pmax,
                               int nb, float* __restrict__ mm)
{
  __shared__ float smn[256], smx[256];
  float mn = 3.4e38f, mx = -3.4e38f;
  for (int i = threadIdx.x; i < nb; i += 256) {
    mn = fminf(mn, pmin[i]); mx = fmaxf(mx, pmax[i]);
  }
  smn[threadIdx.x] = mn; smx[threadIdx.x] = mx; __syncthreads();
  for (int s = 128; s > 0; s >>= 1) {
    if ((int)threadIdx.x < s) {
      smn[threadIdx.x] = fminf(smn[threadIdx.x], smn[threadIdx.x+s]);
      smx[threadIdx.x] = fmaxf(smx[threadIdx.x], smx[threadIdx.x+s]);
    }
    __syncthreads();
  }
  if (threadIdx.x == 0) { mm[0] = smn[0]; mm[1] = smx[0]; }
}

__global__ void k_norm(const float* __restrict__ x, const float* __restrict__ mm,
                       unsigned short* __restrict__ o, int n)
{
  int i = blockIdx.x*blockDim.x + threadIdx.x;
  if (i >= n) return;
  float mn = mm[0], mx = mm[1];
  float den = (mx - mn > 0.f) ? (mx - mn) : 1.f;
  o[i] = f2bf((x[i] - mn) / den);
}

// ===================== instance norm (per channel) + relu -> bf16 ============
__global__ void k_inorm_stats(const float* __restrict__ x, int Nv, float* __restrict__ stats)
{
  int c = blockIdx.x;
  __shared__ float s1[256], s2[256];
  float a = 0.f, b = 0.f;
  const float* xc = x + (size_t)c*Nv;
  for (int i = threadIdx.x; i < Nv; i += 256) { float v = xc[i]; a += v; b += v*v; }
  s1[threadIdx.x] = a; s2[threadIdx.x] = b; __syncthreads();
  for (int s = 128; s > 0; s >>= 1) {
    if ((int)threadIdx.x < s) { s1[threadIdx.x] += s1[threadIdx.x+s]; s2[threadIdx.x] += s2[threadIdx.x+s]; }
    __syncthreads();
  }
  if (threadIdx.x == 0) {
    float mean = s1[0]/Nv;
    float var  = s2[0]/Nv - mean*mean;
    stats[2*c]   = mean;
    stats[2*c+1] = rsqrtf(fmaxf(var, 0.f) + 1e-5f);
  }
}

__global__ void k_inorm_apply(const float* __restrict__ x, const float* __restrict__ stats,
                              unsigned short* __restrict__ o, int Nv, long long total)
{
  long long idx = (long long)blockIdx.x*blockDim.x + threadIdx.x;
  if (idx >= total) return;
  int c = (int)(idx / Nv);
  float v = (x[idx] - stats[2*c]) * stats[2*c+1];
  o[idx] = f2bf(fmaxf(v, 0.f));
}

// ===================== context split: net=tanh, ctx=relu =====================
__global__ void k_split(const float* __restrict__ cm, float* __restrict__ netF,
                        unsigned short* __restrict__ hB, unsigned short* __restrict__ ctxB, int M)
{
  int idx = blockIdx.x*blockDim.x + threadIdx.x;
  if (idx >= 160*M) return;
  int c = idx / M, m = idx - c*M;
  float v = cm[idx];
  if (c < 96) { float t = tanhf(v); netF[c*M+m] = t; hB[c*M+m] = f2bf(t); }
  else        { ctxB[(size_t)(c-96)*M + m] = f2bf(fmaxf(v, 0.f)); }
}

// ===================== pyramid pooling =======================================
__global__ void k_avgpool(const float* __restrict__ in, float* __restrict__ out, int n, int so)
{
  int idx = blockIdx.x*blockDim.x + threadIdx.x;
  int vol = so*so*so;
  if (idx >= n*vol) return;
  int m = idx / vol; int r = idx - m*vol;
  int z = r/(so*so); int r2 = r - z*so*so; int y = r2/so; int x = r2 - y*so;
  int si = so*2;
  const float* base = in + (size_t)m*si*si*si;
  float s = 0.f;
  for (int dz = 0; dz < 2; ++dz)
    for (int dy = 0; dy < 2; ++dy)
      for (int dx = 0; dx < 2; ++dx)
        s += base[((2*z+dz)*si + (2*y+dy))*si + (2*x+dx)];
  out[idx] = s * 0.125f;
}

// ===================== correlation lookup (4 levels, 729 offsets) ============
__device__ __forceinline__ float trisample(const float* __restrict__ vol, int s,
                                           float pz, float py, float px)
{
  int z0 = (int)floorf(pz), y0 = (int)floorf(py), x0 = (int)floorf(px);
  float fz = pz - z0, fy = py - y0, fx = px - x0;
  float acc = 0.f;
#pragma unroll
  for (int cz = 0; cz < 2; ++cz)
#pragma unroll
    for (int cy = 0; cy < 2; ++cy)
#pragma unroll
      for (int cx = 0; cx < 2; ++cx) {
        int iz = z0+cz, iy = y0+cy, ix = x0+cx;
        if ((unsigned)iz < (unsigned)s && (unsigned)iy < (unsigned)s && (unsigned)ix < (unsigned)s) {
          float w = (cz?fz:1.f-fz)*(cy?fy:1.f-fy)*(cx?fx:1.f-fx);
          acc += w * vol[(iz*s + iy)*s + ix];
        }
      }
  return acc;
}

__global__ void k_corr_lookup(const float* __restrict__ p0, const float* __restrict__ p1,
                              const float* __restrict__ p2, const float* __restrict__ p3,
                              const float* __restrict__ c1, unsigned short* __restrict__ cf)
{
  int idx = blockIdx.x*blockDim.x + threadIdx.x;
  if (idx >= NV*KOFF) return;
  int m = idx / KOFF, off = idx - m*KOFF;
  float dz = (float)(off/81)     - 4.f;
  float dy = (float)((off/9)%9)  - 4.f;
  float dx = (float)(off%9)      - 4.f;
  float cz = c1[m], cy = c1[NV+m], cx = c1[2*NV+m];
  const float* pyr[4] = {p0, p1, p2, p3};
  float inv = 1.f;
  for (int l = 0; l < 4; ++l) {
    int s = HT >> l;
    const float* vol = pyr[l] + (size_t)m*s*s*s;
    float v = trisample(vol, s, cz*inv + dz, cy*inv + dy, cx*inv + dx);
    cf[(size_t)(l*KOFF + off)*NV + m] = f2bf(v);
    inv *= 0.5f;
  }
}

// ===================== small elementwise =====================================
__global__ void k_coords_init(float* __restrict__ c1)
{
  int idx = blockIdx.x*blockDim.x + threadIdx.x;
  if (idx >= 3*NV) return;
  int c = idx / NV, m = idx - c*NV;
  float base = (c==0) ? (float)(m/144) : ((c==1) ? (float)((m/12)%12) : (float)(m%12));
  c1[idx] = base;
}

__global__ void k_flow(const float* __restrict__ c1, unsigned short* __restrict__ flow3b,
                       unsigned short* __restrict__ hxFlow)
{
  int idx = blockIdx.x*blockDim.x + threadIdx.x;
  if (idx >= 3*NV) return;
  int c = idx / NV, m = idx - c*NV;
  float base = (c==0) ? (float)(m/144) : ((c==1) ? (float)((m/12)%12) : (float)(m%12));
  unsigned short v = f2bf(c1[idx] - base);
  flow3b[idx] = v; hxFlow[idx] = v;
}

__global__ void k_copy_u16(unsigned short* __restrict__ dst, const unsigned short* __restrict__ src,
                           long long n)
{
  long long i = (long long)blockIdx.x*blockDim.x + threadIdx.x;
  if (i < n) dst[i] = src[i];
}

__global__ void k_mul_bf16(const float* __restrict__ r, const float* __restrict__ h,
                           unsigned short* __restrict__ o, int n)
{
  int i = blockIdx.x*blockDim.x + threadIdx.x;
  if (i < n) o[i] = f2bf(r[i]*h[i]);
}

__global__ void k_gru_combine(const float* __restrict__ z, const float* __restrict__ q,
                              float* __restrict__ h, unsigned short* __restrict__ hB, int n)
{
  int i = blockIdx.x*blockDim.x + threadIdx.x;
  if (i >= n) return;
  float zv = z[i];
  float nv = (1.f - zv)*h[i] + zv*q[i];
  h[i] = nv; hB[i] = f2bf(nv);
}

__global__ void k_add(float* __restrict__ a, const float* __restrict__ b, int n)
{
  int i = blockIdx.x*blockDim.x + threadIdx.x;
  if (i < n) a[i] += b[i];
}

// ===================== flow upsample 12^3 -> 96^3, *8 ========================
__global__ void k_resize_flow(const float* __restrict__ c1, float* __restrict__ out)
{
  int idx = blockIdx.x*blockDim.x + threadIdx.x;
  if (idx >= 3*NFULL) return;
  int c = idx / NFULL; int rem = idx - c*NFULL;
  int z = rem/(HFULL*HFULL); int r2 = rem - z*HFULL*HFULL;
  int y = r2/HFULL; int x = r2 - y*HFULL;
  float sz = (z + 0.5f)*0.125f - 0.5f;
  float sy = (y + 0.5f)*0.125f - 0.5f;
  float sx = (x + 0.5f)*0.125f - 0.5f;
  int z0 = (int)floorf(sz), y0 = (int)floorf(sy), x0 = (int)floorf(sx);
  float fz = sz - z0, fy = sy - y0, fx = sx - x0;
  float acc = 0.f;
#pragma unroll
  for (int cz = 0; cz < 2; ++cz)
#pragma unroll
    for (int cy = 0; cy < 2; ++cy)
#pragma unroll
      for (int cx = 0; cx < 2; ++cx) {
        int iz = min(max(z0+cz, 0), HT-1);
        int iy = min(max(y0+cy, 0), HT-1);
        int ix = min(max(x0+cx, 0), HT-1);
        float base = (c==0) ? (float)iz : ((c==1) ? (float)iy : (float)ix);
        float v = c1[c*NV + (iz*HT + iy)*HT + ix] - base;
        acc += (cz?fz:1.f-fz)*(cy?fy:1.f-fy)*(cx?fx:1.f-fx) * v;
      }
  out[idx] = acc * 8.f;
}

struct WDesc { unsigned short* buf; int Ktot, Kpad, Cout, Npad; int kvol, khw, kw; };

} // namespace raftdvc

// ===================== host orchestration ====================================
extern "C" void kernel_launch(void* const* d_in, const int* in_sizes, int n_in,
                              void* d_out, int out_size, void* d_ws, size_t ws_size,
                              hipStream_t stream)
{
  using namespace raftdvc;
  (void)in_sizes; (void)n_in; (void)out_size; (void)ws_size;

  // ---- inputs (setup_inputs dict order, params in insertion order) ----
  int ii = 0;
  const float* vol0 = (const float*)d_in[ii++];
  const float* vol1 = (const float*)d_in[ii++];
  const float *fw1=(const float*)d_in[ii++], *fb1=(const float*)d_in[ii++];
  const float *fw2=(const float*)d_in[ii++], *fb2=(const float*)d_in[ii++];
  const float *fw3=(const float*)d_in[ii++], *fb3=(const float*)d_in[ii++];
  const float *fw4=(const float*)d_in[ii++], *fb4=(const float*)d_in[ii++];
  const float *cw1=(const float*)d_in[ii++], *cb1=(const float*)d_in[ii++];
  const float *cw2=(const float*)d_in[ii++], *cb2=(const float*)d_in[ii++];
  const float *cw3=(const float*)d_in[ii++], *cb3=(const float*)d_in[ii++];
  const float *cw4=(const float*)d_in[ii++], *cb4=(const float*)d_in[ii++];
  const float *mc1w=(const float*)d_in[ii++], *mc1b=(const float*)d_in[ii++];
  const float *mc2w=(const float*)d_in[ii++], *mc2b=(const float*)d_in[ii++];
  const float *mf1w=(const float*)d_in[ii++], *mf1b=(const float*)d_in[ii++];
  const float *mf2w=(const float*)d_in[ii++], *mf2b=(const float*)d_in[ii++];
  const float *mcfw=(const float*)d_in[ii++], *mcfb=(const float*)d_in[ii++];
  const float *gw[9], *gb[9];
  for (int a = 0; a < 9; ++a) { gw[a]=(const float*)d_in[ii++]; gb[a]=(const float*)d_in[ii++]; }
  const float *fh1w=(const float*)d_in[ii++], *fh1bb=(const float*)d_in[ii++];
  const float *fh2w=(const float*)d_in[ii++], *fh2bb=(const float*)d_in[ii++];

  float* out = (float*)d_out;

  // ---- workspace bump allocator ----
  char* wbase = (char*)d_ws; size_t woff = 0;
  auto alloc = [&](size_t bytes) -> void* {
    woff = (woff + 255) & ~(size_t)255;
    void* p = wbase + woff; woff += bytes; return p;
  };
  auto up32 = [](int x){ return (x + 31) & ~31; };
  auto up16 = [](int x){ return (x + 15) & ~15; };

  // activation / state buffers
  unsigned short* v0b = (unsigned short*)alloc((size_t)NFULL*2);
  unsigned short* v1b = (unsigned short*)alloc((size_t)NFULL*2);
  float* pmin = (float*)alloc(1024*4);
  float* pmax = (float*)alloc(1024*4);
  float* mm   = (float*)alloc(2*4);
  float* stats = (float*)alloc(2*96*4);
  float*          e1f = (float*)alloc((size_t)32*110592*4);
  unsigned short* e1b = (unsigned short*)alloc((size_t)32*110592*2);
  float*          e2f = (float*)alloc((size_t)64*13824*4);
  unsigned short* e2b = (unsigned short*)alloc((size_t)64*13824*2);
  float*          e3f = (float*)alloc((size_t)96*NV*4);
  unsigned short* e3b = (unsigned short*)alloc((size_t)96*NV*2);
  float*          cmf = (float*)alloc((size_t)160*NV*4);
  unsigned short* f0b = (unsigned short*)alloc((size_t)FDIM*NV*2);
  unsigned short* f1b = (unsigned short*)alloc((size_t)FDIM*NV*2);
  float* corr = (float*)alloc((size_t)NV*NV*4);
  float* pyr1 = (float*)alloc((size_t)NV*216*4);
  float* pyr2 = (float*)alloc((size_t)NV*27*4);
  float* pyr3 = (float*)alloc((size_t)NV*1*4);
  float* netF = (float*)alloc((size_t)96*NV*4);
  unsigned short* hx  = (unsigned short*)alloc((size_t)288*NV*2); // [h96|ctx64|mcf125|flow3]
  unsigned short* rhx = (unsigned short*)alloc((size_t)288*NV*2);
  unsigned short* cfeat = (unsigned short*)alloc((size_t)CORR_PLANES*NV*2);
  unsigned short* mc1o  = (unsigned short*)alloc((size_t)192*NV*2);
  unsigned short* cf    = (unsigned short*)alloc((size_t)192*NV*2); // [cor128|flo64]
  unsigned short* flow3b = (unsigned short*)alloc((size_t)3*NV*2);
  unsigned short* mf1o   = (unsigned short*)alloc((size_t)96*NV*2);
  float* zf = (float*)alloc((size_t)96*NV*4);
  float* rf = (float*)alloc((size_t)96*NV*4);
  float* qf = (float*)alloc((size_t)96*NV*4);
  unsigned short* fh1o = (unsigned short*)alloc((size_t)256*NV*2);
  float* dflow   = (float*)alloc((size_t)3*NV*4);
  float* coords1 = (float*)alloc((size_t)3*NV*4);

  // ---- repack all weights to bf16 [Kpad][Npad] ----
  auto repack = [&](const float* w, int Cout, int Cin, int kd, int kh, int kw) -> WDesc {
    WDesc d;
    d.kvol = kd*kh*kw; d.khw = kh*kw; d.kw = kw;
    d.Ktot = Cin*d.kvol; d.Kpad = up32(d.Ktot); d.Cout = Cout; d.Npad = up16(Cout);
    d.buf  = (unsigned short*)alloc((size_t)d.Kpad*d.Npad*2);
    size_t tot = (size_t)d.Kpad*d.Npad;
    k_repack<<<dim3((unsigned)((tot + 255)/256)), 256, 0, stream>>>(
        w, d.buf, Cout, Cin, kd, kh, kw, d.Kpad, d.Npad);
    return d;
  };

  WDesc Wf1 = repack(fw1, 32, 1, 7,7,7),   Wf2 = repack(fw2, 64, 32, 3,3,3);
  WDesc Wf3 = repack(fw3, 96, 64, 3,3,3),  Wf4 = repack(fw4, FDIM, 96, 1,1,1);
  WDesc Wc1 = repack(cw1, 32, 1, 7,7,7),   Wc2 = repack(cw2, 64, 32, 3,3,3);
  WDesc Wc3 = repack(cw3, 96, 64, 3,3,3),  Wc4 = repack(cw4, 160, 96, 1,1,1);
  WDesc Wmc1 = repack(mc1w, 192, CORR_PLANES, 1,1,1);
  WDesc Wmc2 = repack(mc2w, 128, 192, 3,3,3);
  WDesc Wmf1 = repack(mf1w, 96, 3, 3,3,3);
  WDesc Wmf2 = repack(mf2w, 64, 96, 3,3,3);
  WDesc Wmcf = repack(mcfw, 125, 192, 3,3,3);
  const int gk[3][3] = {{5,1,1},{1,5,1},{1,1,5}};
  const int gp[3][3] = {{2,0,0},{0,2,0},{0,0,2}};
  WDesc Wg[9];
  for (int a = 0; a < 3; ++a)
    for (int g = 0; g < 3; ++g)
      Wg[a*3+g] = repack(gw[a*3+g], 96, 288, gk[a][0], gk[a][1], gk[a][2]);
  WDesc Wfh1 = repack(fh1w, 256, 96, 3,3,3);
  WDesc Wfh2 = repack(fh2w, 3, 256, 3,3,3);

  // ---- conv launcher ----
  auto conv = [&](const unsigned short* A, const WDesc& wd, const float* bias,
                  float* outFp, unsigned short* outBp,
                  int Do, int Ho, int Wo, int Di, int Hi, int Wi,
                  int s, int pz, int py, int px,
                  float scale, int act) {
    int M = Do*Ho*Wo;
    dim3 g((unsigned)((M/16 + 3)/4), (unsigned)((wd.Npad + 31)/32));
    k_conv3d_wmma<<<g, 128, 0, stream>>>(A, wd.buf, bias, outFp, outBp,
        M, Ho, Wo, Di, Hi, Wi, wd.kvol, wd.khw, wd.kw, s, pz, py, px,
        wd.Ktot, wd.Kpad, wd.Cout, wd.Npad, scale, act);
  };
  auto inorm_relu = [&](const float* x, unsigned short* o, int C, int Nvx) {
    k_inorm_stats<<<C, 256, 0, stream>>>(x, Nvx, stats);
    long long tot = (long long)C*Nvx;
    k_inorm_apply<<<dim3((unsigned)((tot + 255)/256)), 256, 0, stream>>>(x, stats, o, Nvx, tot);
  };

  // ---- normalize inputs ----
  k_minmax_part<<<1024, 256, 0, stream>>>(vol0, vol1, NFULL, pmin, pmax);
  k_minmax_final<<<1, 256, 0, stream>>>(pmin, pmax, 1024, mm);
  k_norm<<<(NFULL + 255)/256, 256, 0, stream>>>(vol0, mm, v0b, NFULL);
  k_norm<<<(NFULL + 255)/256, 256, 0, stream>>>(vol1, mm, v1b, NFULL);

  // ---- encoders ----
  auto encoder = [&](const unsigned short* inB, const WDesc& w1, const WDesc& w2,
                     const WDesc& w3, const WDesc& w4,
                     const float* b1, const float* b2, const float* b3, const float* b4,
                     float* outFp, unsigned short* outBp) {
    conv(inB, w1, b1, e1f, nullptr, 48,48,48, 96,96,96, 2, 2,2,2, 1.f, ACT_NONE);
    inorm_relu(e1f, e1b, 32, 48*48*48);
    conv(e1b, w2, b2, e2f, nullptr, 24,24,24, 48,48,48, 2, 0,0,0, 1.f, ACT_NONE);
    inorm_relu(e2f, e2b, 64, 24*24*24);
    conv(e2b, w3, b3, e3f, nullptr, 12,12,12, 24,24,24, 2, 0,0,0, 1.f, ACT_NONE);
    inorm_relu(e3f, e3b, 96, NV);
    conv(e3b, w4, b4, outFp, outBp, 12,12,12, 12,12,12, 1, 0,0,0, 1.f, ACT_NONE);
  };
  encoder(v0b, Wf1, Wf2, Wf3, Wf4, fb1, fb2, fb3, fb4, nullptr, f0b);
  encoder(v1b, Wf1, Wf2, Wf3, Wf4, fb1, fb2, fb3, fb4, nullptr, f1b);
  encoder(v0b, Wc1, Wc2, Wc3, Wc4, cb1, cb2, cb3, cb4, cmf, nullptr);

  // ---- correlation volume: corr[q][t] = <f0[:,q], f1[:,t]> / sqrt(128) ----
  WDesc Wcorr; Wcorr.buf = f0b; Wcorr.Ktot = FDIM; Wcorr.Kpad = FDIM;
  Wcorr.Cout = NV; Wcorr.Npad = NV; Wcorr.kvol = 1; Wcorr.khw = 1; Wcorr.kw = 1;
  conv(f1b, Wcorr, nullptr, corr, nullptr, 12,12,12, 12,12,12, 1, 0,0,0,
       0.08838834764831845f, ACT_NONE);

  // ---- pyramid ----
  k_avgpool<<<(NV*216 + 255)/256, 256, 0, stream>>>(corr, pyr1, NV, 6);
  k_avgpool<<<(NV*27  + 255)/256, 256, 0, stream>>>(pyr1, pyr2, NV, 3);
  k_avgpool<<<(NV*1   + 255)/256, 256, 0, stream>>>(pyr2, pyr3, NV, 1);

  // ---- context split (net->h region of hx, ctx->x region of hx) ----
  k_split<<<(160*NV + 255)/256, 256, 0, stream>>>(cmf, netF, hx, hx + (size_t)96*NV, NV);
  k_coords_init<<<(3*NV + 255)/256, 256, 0, stream>>>(coords1);

  // ---- iterations ----
  for (int it = 0; it < ITERS; ++it) {
    k_corr_lookup<<<(NV*KOFF + 255)/256, 256, 0, stream>>>(corr, pyr1, pyr2, pyr3,
                                                           coords1, cfeat);
    k_flow<<<(3*NV + 255)/256, 256, 0, stream>>>(coords1, flow3b, hx + (size_t)285*NV);

    // motion encoder
    conv(cfeat, Wmc1, mc1b, nullptr, mc1o, 12,12,12, 12,12,12, 1, 0,0,0, 1.f, ACT_RELU);
    conv(mc1o,  Wmc2, mc2b, nullptr, cf,   12,12,12, 12,12,12, 1, 1,1,1, 1.f, ACT_RELU);
    conv(flow3b, Wmf1, mf1b, nullptr, mf1o, 12,12,12, 12,12,12, 1, 1,1,1, 1.f, ACT_RELU);
    conv(mf1o,  Wmf2, mf2b, nullptr, cf + (size_t)128*NV, 12,12,12, 12,12,12, 1, 1,1,1, 1.f, ACT_RELU);
    conv(cf,    Wmcf, mcfb, nullptr, hx + (size_t)160*NV, 12,12,12, 12,12,12, 1, 1,1,1, 1.f, ACT_RELU);
    // mirror the x region (ctx|mot) into rhx
    k_copy_u16<<<(192*NV + 255)/256, 256, 0, stream>>>(rhx + (size_t)96*NV,
                                                       hx + (size_t)96*NV, (long long)192*NV);

    // 3x GRU
    for (int a = 0; a < 3; ++a) {
      conv(hx, Wg[a*3+0], gb[a*3+0], zf, nullptr, 12,12,12, 12,12,12,
           1, gp[a][0], gp[a][1], gp[a][2], 1.f, ACT_SIGMOID);
      conv(hx, Wg[a*3+1], gb[a*3+1], rf, nullptr, 12,12,12, 12,12,12,
           1, gp[a][0], gp[a][1], gp[a][2], 1.f, ACT_SIGMOID);
      k_mul_bf16<<<(96*NV + 255)/256, 256, 0, stream>>>(rf, netF, rhx, 96*NV);
      conv(rhx, Wg[a*3+2], gb[a*3+2], qf, nullptr, 12,12,12, 12,12,12,
           1, gp[a][0], gp[a][1], gp[a][2], 1.f, ACT_TANH);
      k_gru_combine<<<(96*NV + 255)/256, 256, 0, stream>>>(zf, qf, netF, hx, 96*NV);
    }

    // flow head
    conv(hx, Wfh1, fh1bb, nullptr, fh1o, 12,12,12, 12,12,12, 1, 1,1,1, 1.f, ACT_RELU);
    conv(fh1o, Wfh2, fh2bb, dflow, nullptr, 12,12,12, 12,12,12, 1, 1,1,1, 1.f, ACT_NONE);
    k_add<<<(3*NV + 255)/256, 256, 0, stream>>>(coords1, dflow, 3*NV);
    k_resize_flow<<<(3*NFULL + 255)/256, 256, 0, stream>>>(coords1,
                                                           out + (size_t)it*3*NFULL);
  }
}
